// PeakPreservingAttention_31361851195619
// MI455X (gfx1250) — matrix operand order repaired and checked by reading, not compiled, not run on hardware
//
#include <hip/hip_runtime.h>

typedef __attribute__((ext_vector_type(16))) _Float16     v16h;
typedef __attribute__((ext_vector_type(8)))  float        v8f;
typedef __attribute__((ext_vector_type(4)))  unsigned int u32x4;
typedef __attribute__((ext_vector_type(4)))  int          i32x4;
typedef __attribute__((ext_vector_type(8)))  int          i32x8;

#define D  64
#define DH 32
#define NEG_SLOPE 0.2f

// ---------------- TDM: 2D f32 tile load global -> LDS ----------------
// Builds a D# per CDNA5 ISA and issues TENSOR_LOAD_TO_LDS.
// width = row length (elements) = tensor_dim0 = tile_dim0 = tensor_dim0_stride.
// tensor_rows = rows available from tile start (OOB rows read as zero).
__device__ __forceinline__ void tdm_load_tile_f32(const float* g, unsigned lds_off,
                                                  unsigned tensor_rows,
                                                  unsigned tile_rows,
                                                  unsigned width) {
  unsigned long long ga = (unsigned long long)(uintptr_t)g;
  u32x4 g0 = {
    1u,                                                   // count=1 valid descriptor
    lds_off,                                              // LDS byte address
    (unsigned)(ga & 0xffffffffu),                         // global_addr[31:0]
    (unsigned)((ga >> 32) & 0x01ffffffu) | (2u << 30)     // global_addr[56:32] | type=2
  };
  i32x8 g1 = {
    (int)(2u << 16),                                      // data_size=2 -> 4 bytes
    (int)((width & 0xffffu) << 16),                       // tensor_dim0[15:0]
    (int)(((width >> 16) & 0xffffu) |
          ((tensor_rows & 0xffffu) << 16)),               // dim0[31:16] | dim1[15:0]
    (int)(((tensor_rows >> 16) & 0xffffu) |
          ((width & 0xffffu) << 16)),                     // dim1[31:16] | tile_dim0
    (int)(tile_rows & 0xffffu),                           // tile_dim1 (tile_dim2=0)
    (int)width,                                           // tensor_dim0_stride[31:0]
    0,                                                    // stride0 hi | dim1_stride lo
    0
  };
  i32x4 z4 = {0, 0, 0, 0};
#if defined(__clang_major__) && __clang_major__ >= 23
  i32x8 z8 = {0, 0, 0, 0, 0, 0, 0, 0};
  __builtin_amdgcn_tensor_load_to_lds(g0, g1, z4, z4, z8, 0);
#else
  __builtin_amdgcn_tensor_load_to_lds(g0, g1, z4, z4, 0);
#endif
}

// ---------------- float atomic max via sign-aware punning ----------------
__device__ inline void atomicMaxF(float* addr, float val) {
  if (val >= 0.0f) atomicMax((int*)addr, __float_as_int(val));
  else             atomicMin((unsigned int*)addr, __float_as_uint(val));
}

// ---------------- kernel 0: init softmax stats ----------------
__global__ __launch_bounds__(256) void k_init(float* __restrict__ m, float* __restrict__ s, int N) {
  int i = blockIdx.x * 256 + threadIdx.x;
  if (i < N) { m[i] = -INFINITY; s[i] = 0.0f; }
}

// ---------------- kernel 1: h = x @ W_lin via WMMA (f16 in, f32 acc) ----------------
// 256 threads = 8 waves; each wave computes a 16x64 tile -> 128 rows/block.
// x tile and W are DMA'd into LDS by the Tensor Data Mover (one descriptor each).
__global__ __launch_bounds__(256) void k_proj(const float* __restrict__ x,
                                              const float* __restrict__ W,
                                              float* __restrict__ h, int N) {
  __shared__ float sW[D * D];     // 16 KB
  __shared__ float sX[128 * D];   // 32 KB
  const int tid = threadIdx.x;
  const int rowBase = blockIdx.x * 128;

  // Wave 0 DMAs the 128x64 x tile; wave 1 DMAs W. TDM ignores EXEC; one issue/wave.
  if (tid < 64) {
    if (tid < 32) {
      tdm_load_tile_f32(x + (size_t)rowBase * D, (unsigned)(uintptr_t)&sX[0],
                        (unsigned)(N - rowBase), 128u, (unsigned)D);
    } else {
      tdm_load_tile_f32(W, (unsigned)(uintptr_t)&sW[0], (unsigned)D, (unsigned)D,
                        (unsigned)D);
    }
    __builtin_amdgcn_s_wait_tensorcnt(0);
  }
  __syncthreads();

  const int wave = tid >> 5;
  const int lane = tid & 31;
  const int lm   = lane & 15;   // M for A, N for B/C
  const int lh   = lane >> 4;   // half selector

  // A fragments: 16x32 f16, lane holds M=lm
  // (VGPR0-3: K=(lh?8:0)+0..7 ; VGPR4-7: K=16+(lh?8:0)+0..7)
  const int arow = wave * 16 + lm;
  v16h A[2];
#pragma unroll
  for (int kt = 0; kt < 2; ++kt) {
    union { v16h v; _Float16 e[16]; } ua;
    const int kb = kt * 32 + lh * 8;
#pragma unroll
    for (int i = 0; i < 8; ++i) {
      ua.e[i]     = (_Float16)sX[arow * D + kb + i];
      ua.e[8 + i] = (_Float16)sX[arow * D + kb + 16 + i];
    }
    A[kt] = ua.v;
  }

  const bool full = (rowBase + 128) <= N;   // uniform per block

#pragma unroll
  for (int nt = 0; nt < 4; ++nt) {
    v8f acc = {};
#pragma unroll
    for (int kt = 0; kt < 2; ++kt) {
      // B fragment 32x16 f16: N=lm; e0-7: K=(lh?8:0)+0..7 ; e8-15: K=16+(lh?8:0)+0..7
      union { v16h v; _Float16 e[16]; } ub;
      const int kb = kt * 32 + lh * 8;
      const int n  = nt * 16 + lm;
#pragma unroll
      for (int i = 0; i < 16; ++i)
        ub.e[i] = (_Float16)sW[(kb + i + (i >= 8 ? 8 : 0)) * D + n];
      acc = __builtin_amdgcn_wmma_f32_16x16x32_f16(false, A[kt], false, ub.v,
                                                   (short)0, acc, false, false);
    }
    // C/D layout: VGPR v -> M = v + lh*8, N = lm
    union { v8f v; float e[8]; } uc; uc.v = acc;
    const int col = nt * 16 + lm;
    if (full) {
#pragma unroll
      for (int v = 0; v < 8; ++v)
        h[(size_t)(rowBase + wave * 16 + lh * 8 + v) * D + col] = uc.e[v];
    } else {
#pragma unroll
      for (int v = 0; v < 8; ++v) {
        const int grow = rowBase + wave * 16 + lh * 8 + v;
        if (grow < N) h[(size_t)grow * D + col] = uc.e[v];
      }
    }
  }
}

// ---------------- kernel 2: per-node scalars (a_s, a_d, peak MLP) ----------------
__global__ __launch_bounds__(256) void k_node(const float* __restrict__ h,
    const float* __restrict__ att_src, const float* __restrict__ att_dst,
    const float* __restrict__ W1, const float* __restrict__ b1,
    const float* __restrict__ W2, const float* __restrict__ b2,
    float* __restrict__ a_s, float* __restrict__ a_d, float* __restrict__ peak,
    int N) {
  __shared__ float sH[64 * D];    // 16 KB
  __shared__ float sW1[D * DH];   // 8 KB
  __shared__ float sAs[D], sAd[D], sB1[DH], sW2[DH];
  __shared__ float rS[256], rDp[256], rL[256];
  const int tid  = threadIdx.x;
  const int base = blockIdx.x * 64;

  for (int i = tid * 4; i < 64 * D; i += 1024) {
    int r = base + (i >> 6);
    if (r >= N) r = N - 1;
    *(float4*)&sH[i] = *(const float4*)&h[r * D + (i & 63)];
  }
  for (int i = tid * 4; i < D * DH; i += 1024)
    *(float4*)&sW1[i] = *(const float4*)&W1[i];
  if (tid < D)  { sAs[tid] = att_src[tid]; sAd[tid] = att_dst[tid]; }
  if (tid < DH) { sB1[tid] = b1[tid]; sW2[tid] = W2[tid]; }
  __syncthreads();

  const int nl   = tid >> 2;
  const int part = tid & 3;
  const float* hr = &sH[nl * D];

  float ps = 0.f, pd = 0.f;
  for (int k = part * 16; k < part * 16 + 16; ++k) {
    float v = hr[k];
    ps += v * sAs[k];
    pd += v * sAd[k];
  }
  float pl = 0.f;
  for (int j = part * 8; j < part * 8 + 8; ++j) {
    float acc = sB1[j];
    for (int k = 0; k < D; ++k) acc += hr[k] * sW1[k * DH + j];
    float g = 0.5f * acc * (1.0f + erff(acc * 0.70710678f)); // exact gelu
    pl += g * sW2[j];
  }
  rS[tid] = ps; rDp[tid] = pd; rL[tid] = pl;
  __syncthreads();
  if (part == 0) {
    int node = base + nl;
    if (node < N) {
      float as = rS[tid] + rS[tid + 1] + rS[tid + 2] + rS[tid + 3];
      float ad = rDp[tid] + rDp[tid + 1] + rDp[tid + 2] + rDp[tid + 3];
      float lg = rL[tid] + rL[tid + 1] + rL[tid + 2] + rL[tid + 3] + b2[0];
      a_s[node]  = as;
      a_d[node]  = ad;
      peak[node] = 1.0f / (1.0f + expf(-lg));
    }
  }
}

// ---------------- kernel 3: edge pass 1 — alpha + segment max ----------------
__global__ __launch_bounds__(256) void k_edge_max(const long long* __restrict__ ei,
    const float* __restrict__ a_s, const float* __restrict__ a_d,
    float* __restrict__ alpha, float* __restrict__ m, long long E) {
  long long e = (long long)blockIdx.x * 256 + threadIdx.x;
  if (e >= E) return;
  int r = (int)ei[e];
  int c = (int)ei[E + e];
  float a = a_s[r] + a_d[c];
  a = (a >= 0.f) ? a : NEG_SLOPE * a;
  alpha[e] = a;
  atomicMaxF(&m[r], a);
}

// ---------------- kernel 4: edge pass 2 — exp + segment sum ----------------
__global__ __launch_bounds__(256) void k_edge_sum(const long long* __restrict__ ei,
    const float* __restrict__ alpha, const float* __restrict__ m,
    float* __restrict__ s, long long E) {
  long long e = (long long)blockIdx.x * 256 + threadIdx.x;
  if (e >= E) return;
  int r = (int)ei[e];
  float v = expf(alpha[e] - m[r]);
  atomicAdd(&s[r], v);
}

// ---------------- kernel 5: finalize ----------------
// segment_sum(h[row]*alpha, row) == h[n] * sum(alpha_norm) == h[n] for nodes with
// out-edges (softmax sums to 1 per segment), 0 for empty segments. Exact refactor.
__global__ __launch_bounds__(256) void k_final(const float* __restrict__ h,
    const float* __restrict__ s, const float* __restrict__ peak,
    const float* __restrict__ bias, float* __restrict__ out, int N) {
  int i = blockIdx.x * 256 + threadIdx.x;
  if (i >= N * D) return;
  int n = i >> 6, d = i & 63;
  float scale = (s[n] > 0.f ? 1.f : 0.f) * (1.f + peak[n]);
  out[i] = h[i] * scale + bias[d];
}

extern "C" void kernel_launch(void* const* d_in, const int* in_sizes, int n_in,
                              void* d_out, int out_size, void* d_ws, size_t ws_size,
                              hipStream_t stream) {
  const float*     x       = (const float*)d_in[0];
  const long long* ei      = (const long long*)d_in[1];   // int64 edge_index [2,E]
  const float*     W_lin   = (const float*)d_in[2];
  const float*     att_src = (const float*)d_in[3];
  const float*     att_dst = (const float*)d_in[4];
  const float*     bias    = (const float*)d_in[5];
  const float*     W1      = (const float*)d_in[6];
  const float*     b1      = (const float*)d_in[7];
  const float*     W2      = (const float*)d_in[8];
  const float*     b2      = (const float*)d_in[9];

  const int       N = in_sizes[0] / D;
  const long long E = in_sizes[1] / 2;

  // workspace carve (all f32)
  char* ws = (char*)d_ws;
  size_t off = 0;
  float* h     = (float*)(ws + off); off += (size_t)N * D * sizeof(float);
  float* a_s   = (float*)(ws + off); off += (size_t)N * sizeof(float);
  float* a_d   = (float*)(ws + off); off += (size_t)N * sizeof(float);
  float* peak  = (float*)(ws + off); off += (size_t)N * sizeof(float);
  float* m     = (float*)(ws + off); off += (size_t)N * sizeof(float);
  float* s     = (float*)(ws + off); off += (size_t)N * sizeof(float);
  float* alpha = (float*)(ws + off); off += (size_t)E * sizeof(float);

  float* out = (float*)d_out;

  k_init<<<(N + 255) / 256, 256, 0, stream>>>(m, s, N);
  k_proj<<<(N + 127) / 128, 256, 0, stream>>>(x, W_lin, h, N);
  k_node<<<(N + 63) / 64, 256, 0, stream>>>(h, att_src, att_dst, W1, b1, W2, b2,
                                            a_s, a_d, peak, N);
  k_edge_max<<<(unsigned)((E + 255) / 256), 256, 0, stream>>>(ei, a_s, a_d, alpha, m, E);
  k_edge_sum<<<(unsigned)((E + 255) / 256), 256, 0, stream>>>(ei, alpha, m, s, E);
  k_final<<<(N * D + 255) / 256, 256, 0, stream>>>(h, s, peak, bias, out, N);
}
